// GateAggregator_21964462751908
// MI455X (gfx1250) — compile-verified, matrix operand order checked by reading, not run on hardware
//
#include <hip/hip_runtime.h>

typedef __attribute__((ext_vector_type(16))) _Float16 v16h;
typedef __attribute__((ext_vector_type(8)))  float    v8f;

namespace {
constexpr int kBatch = 16, kChan = 64, kWin = 200, kPatch = 30;
constexpr int kHid = 128, kGates = 512, kKtot = 192;   // 192 = 64 (x) + 128 (h)
constexpr int kSeq = kBatch * kPatch;                  // 480 sequences

// padded LDS strides (in elements) to avoid bank conflicts on 64 banks
constexpr int WSTR = 200;   // weight row stride (halfs), 192 + 8
constexpr int HSTR = 136;   // h row stride (halfs), 128 + 8
constexpr int XSTR = 72;    // x row stride (halfs), 64 + 8
constexpr int H32S = 132;   // fp32 h row stride (floats)

constexpr size_t OFF_W   = 0;
constexpr size_t SZ_W    = size_t(kGates) * WSTR * 2;   // 204800 B  f16 [W_ih | W_hh]
constexpr size_t OFF_B   = OFF_W + SZ_W;
constexpr size_t SZ_B    = size_t(kGates) * 4;          //   2048 B  f32 bias (b_ih+b_hh)
constexpr size_t OFF_H   = OFF_B + SZ_B;
constexpr size_t SZ_H    = size_t(2) * 16 * HSTR * 2;   //   8704 B  f16 h state, DOUBLE buffered
constexpr size_t OFF_X   = OFF_H + SZ_H;
constexpr size_t SZ_X    = size_t(2) * 16 * XSTR * 2;   //   4608 B  f16 x double buffer
constexpr size_t OFF_H32 = OFF_X + SZ_X;
constexpr size_t SZ_H32  = size_t(16) * H32S * 4;       //   8448 B  f32 final h
constexpr size_t SMEM_BYTES = OFF_H32 + SZ_H32;         // ~229 KB of the 320 KB WGP LDS
}

__device__ __forceinline__ float fast_tanh(float x) {
#if __has_builtin(__builtin_amdgcn_tanhf)
    return __builtin_amdgcn_tanhf(x);           // CDNA5 v_tanh_f32
#elif __has_builtin(__builtin_amdgcn_tanh_f32)
    return __builtin_amdgcn_tanh_f32(x);
#else
    x = fminf(fmaxf(x, -15.0f), 15.0f);
    float e = __expf(-2.0f * x);
    return (1.0f - e) / (1.0f + e);
#endif
}
__device__ __forceinline__ float fast_sigmoid(float x) {
    return 0.5f + 0.5f * fast_tanh(0.5f * x);
}

__global__ __launch_bounds__(256, 1)
void GateAggregator_lstm_kernel(const float* __restrict__ x,
                                const float* __restrict__ W_ih,
                                const float* __restrict__ W_hh,
                                const float* __restrict__ b_ih,
                                const float* __restrict__ b_hh,
                                const float* __restrict__ W_fc,
                                const float* __restrict__ b_fc,
                                float* __restrict__ out)
{
    extern __shared__ char smem[];
    _Float16* sW   = reinterpret_cast<_Float16*>(smem + OFF_W);
    float*    sB   = reinterpret_cast<float*>(smem + OFF_B);
    _Float16* sH   = reinterpret_cast<_Float16*>(smem + OFF_H);    // 2 buffers
    _Float16* sX   = reinterpret_cast<_Float16*>(smem + OFF_X);    // 2 buffers
    float*    sH32 = reinterpret_cast<float*>(smem + OFF_H32);

    const int tid   = threadIdx.x;
    const int lane  = tid & 31;
    const int wave  = tid >> 5;          // 0..7, owns hidden dims [16w, 16w+16)
    const int laneM = lane & 15;
    const int hi8   = (lane >> 4) * 8;   // K offset for upper half-wave; also M offset of D rows

    // ---- one-time: weights -> LDS f16, concatenated per gate row: [W_ih(64) | W_hh(128)]
    for (int i = tid; i < kGates * kKtot; i += 256) {
        int r = i / kKtot, k = i - r * kKtot;
        float v = (k < kChan) ? W_ih[r * kChan + k] : W_hh[r * kHid + (k - kChan)];
        sW[r * WSTR + k] = (_Float16)v;
    }
    for (int i = tid; i < kGates; i += 256) sB[i] = b_ih[i] + b_hh[i];
    for (int i = tid; i < 16 * HSTR; i += 256) sH[i] = (_Float16)0.0f;  // buffer 0 only

    // ---- x staging map: thread handles sequence m=tid&15, channel group cq=tid>>4 (4 channels)
    const int s0 = blockIdx.x * 16;
    const int xm = tid & 15;
    const int cq = tid >> 4;
    const int sq = s0 + xm;
    const int bb = sq / kPatch;
    const int pp = sq - bb * kPatch;
    int xbase[4];
    #pragma unroll
    for (int j = 0; j < 4; ++j) {
        int c = cq + 16 * j;
        xbase[j] = ((bb * kChan + c) * kWin) * kPatch + pp;   // x[b][c][t][p], t stride = kPatch
    }
    #pragma unroll
    for (int j = 0; j < 4; ++j)
        sX[xm * XSTR + (cq + 16 * j)] = (_Float16)x[xbase[j]];   // t = 0, buffer 0
    __syncthreads();

    const int d0 = wave * 16 + laneM;          // my gate column within each 128-wide gate block
    const float bias_i = sB[0 * kHid + d0];
    const float bias_f = sB[1 * kHid + d0];
    const float bias_g = sB[2 * kHid + d0];
    const float bias_o = sB[3 * kHid + d0];

    // ---- hoist ALL loop-invariant B-fragments into registers: 4 gates x 6 k-steps
    // (ks 0..1 cover the x projection W_ih, ks 2..5 cover the recurrence W_hh)
    // B-matrix 16-bit layout: lane = column N (gate row), paired-K dwords, hi-half K+8.
    v16h Ball[4][6];
    #pragma unroll
    for (int q = 0; q < 4; ++q) {
        #pragma unroll
        for (int ks = 0; ks < 6; ++ks) {
            union { v16h v; unsigned u[8]; } F;
            const _Float16* bbase = sW + (q * kHid + d0) * WSTR + 32 * ks;
            #pragma unroll
            for (int r = 0; r < 8; ++r) {
                int k0 = ((r < 4) ? 0 : 16) + hi8 + (r & 3) * 2;
                F.u[r] = *reinterpret_cast<const unsigned*>(bbase + k0);
            }
            Ball[q][ks] = F.v;
        }
    }

    v8f cst;                                   // cell state, register-resident
    #pragma unroll
    for (int j = 0; j < 8; ++j) cst[j] = 0.0f;
    v8f hnew;
    #pragma unroll
    for (int j = 0; j < 8; ++j) hnew[j] = 0.0f;

    int cur = 0;
    for (int t = 0; t < kWin; ++t) {
        // stage x_{t+1} into registers (overlaps with WMMA phase); prefetch t+2
        float xn[4];
        const int tn = (t + 1 < kWin) ? (t + 1) : (kWin - 1);
        #pragma unroll
        for (int j = 0; j < 4; ++j) xn[j] = x[xbase[j] + tn * kPatch];
        if (t + 2 < kWin) {
            #pragma unroll
            for (int j = 0; j < 4; ++j)
                __builtin_prefetch(&x[xbase[j] + (t + 2) * kPatch], 0, 0);
        }

        // ---- preload ALL six A fragments up front, then a pure 24-WMMA burst
        const _Float16* aX = sX + cur * 16 * XSTR + laneM * XSTR;
        const _Float16* aH = sH + cur * 16 * HSTR + laneM * HSTR;
        v16h Af[6];
        #pragma unroll
        for (int ks = 0; ks < 2; ++ks) {
            union { v16h v; unsigned u[8]; } T;
            #pragma unroll
            for (int r = 0; r < 8; ++r) {
                int k0 = ((r < 4) ? 0 : 16) + hi8 + (r & 3) * 2;
                T.u[r] = *reinterpret_cast<const unsigned*>(aX + 32 * ks + k0);
            }
            Af[ks] = T.v;
        }
        #pragma unroll
        for (int ks = 0; ks < 4; ++ks) {
            union { v16h v; unsigned u[8]; } T;
            #pragma unroll
            for (int r = 0; r < 8; ++r) {
                int k0 = ((r < 4) ? 0 : 16) + hi8 + (r & 3) * 2;
                T.u[r] = *reinterpret_cast<const unsigned*>(aH + 32 * ks + k0);
            }
            Af[2 + ks] = T.v;
        }

        // split accumulators: x-chain (depth 2) and h-chain (depth 4) run concurrently
        v8f accX[4], accH[4];
        #pragma unroll
        for (int q = 0; q < 4; ++q)
            #pragma unroll
            for (int j = 0; j < 8; ++j) { accX[q][j] = 0.0f; accH[q][j] = 0.0f; }

        // 24 WMMAs, no LDS in between; 8 independent accumulation chains interleaved
        #pragma unroll
        for (int ks = 0; ks < 2; ++ks)
            #pragma unroll
            for (int q = 0; q < 4; ++q)
                accX[q] = __builtin_amdgcn_wmma_f32_16x16x32_f16(
                    false, Af[ks], false, Ball[q][ks], (short)0, accX[q], false, false);
        #pragma unroll
        for (int ks = 2; ks < 6; ++ks)
            #pragma unroll
            for (int q = 0; q < 4; ++q)
                accH[q] = __builtin_amdgcn_wmma_f32_16x16x32_f16(
                    false, Af[ks], false, Ball[q][ks], (short)0, accH[q], false, false);

        const int nxt = cur ^ 1;

        // commit x_{t+1} into the other buffer (no barrier needed: disjoint buffer)
        #pragma unroll
        for (int j = 0; j < 4; ++j)
            sX[nxt * 16 * XSTR + xm * XSTR + (cq + 16 * j)] = (_Float16)xn[j];

        // LSTM cell update, fully in registers (accumulator layout == c layout);
        // the accX+accH+bias sums dual-issue into the WMMA->VALU hazard window
        #pragma unroll
        for (int j = 0; j < 8; ++j) {
            float gi = fast_sigmoid(accX[0][j] + accH[0][j] + bias_i);
            float gf = fast_sigmoid(accX[1][j] + accH[1][j] + bias_f);
            float gg = fast_tanh   (accX[2][j] + accH[2][j] + bias_g);
            float go = fast_sigmoid(accX[3][j] + accH[3][j] + bias_o);
            float c2 = gf * cst[j] + gi * gg;
            cst[j]  = c2;
            hnew[j] = go * fast_tanh(c2);
        }

        // publish h (f16) into the other h buffer: lane holds rows m = hi8+j, column d0
        #pragma unroll
        for (int j = 0; j < 8; ++j)
            sH[nxt * 16 * HSTR + (hi8 + j) * HSTR + d0] = (_Float16)hnew[j];

        __syncthreads();   // single barrier per step: writes(t) visible before reads(t+1)
        cur = nxt;
    }

    // fp32 final h (hnew persists in registers from the last iteration)
    #pragma unroll
    for (int j = 0; j < 8; ++j)
        sH32[(hi8 + j) * H32S + d0] = hnew[j];
    __syncthreads();

    // ---- final FC: out[m][c] = h[m] . W_fc[c] + b_fc[c]   (tiny; 4 outputs/thread)
    #pragma unroll
    for (int e = 0; e < 4; ++e) {
        int idx = tid * 4 + e;            // 0..1023
        int m   = idx >> 6;               // 0..15
        int cc  = idx & 63;
        const float* wrow = W_fc + cc * kHid;
        const float* hrow = sH32 + m * H32S;
        float sum = b_fc[cc];
        #pragma unroll 8
        for (int k = 0; k < kHid; ++k) sum += hrow[k] * wrow[k];
        out[(s0 + m) * kChan + cc] = sum;
    }
}

extern "C" void kernel_launch(void* const* d_in, const int* in_sizes, int n_in,
                              void* d_out, int out_size, void* d_ws, size_t ws_size,
                              hipStream_t stream) {
    const float* x    = (const float*)d_in[0];
    const float* W_ih = (const float*)d_in[1];
    const float* W_hh = (const float*)d_in[2];
    const float* b_ih = (const float*)d_in[3];
    const float* b_hh = (const float*)d_in[4];
    const float* W_fc = (const float*)d_in[5];
    const float* b_fc = (const float*)d_in[6];
    float* out = (float*)d_out;

    // allow >64KB dynamic LDS (CDNA5 WGP has 320KB)
    hipFuncSetAttribute(reinterpret_cast<const void*>(GateAggregator_lstm_kernel),
                        hipFuncAttributeMaxDynamicSharedMemorySize, (int)SMEM_BYTES);

    GateAggregator_lstm_kernel<<<kSeq / 16, 256, SMEM_BYTES, stream>>>(
        x, W_ih, W_hh, b_ih, b_hh, W_fc, b_fc, out);
}
